// KnotTransformerLayer_6141803233390
// MI455X (gfx1250) — matrix operand order repaired and checked by reading, not compiled, not run on hardware
//
#include <hip/hip_runtime.h>
#include <stdint.h>

#define NN   100000
#define DD   128
#define DFFN 512
#define EPSL 1e-5f
#define QK_SCALE 0.17677669529663687f   // 1/sqrt(32)

typedef __attribute__((ext_vector_type(16))) __bf16         v16bf;
typedef __attribute__((ext_vector_type(8)))  float          v8f;
typedef __attribute__((ext_vector_type(16))) unsigned short v16u;
typedef __attribute__((ext_vector_type(4)))  unsigned int   v4u;

static __device__ __forceinline__ unsigned short f2bf_bits(float f) {
  unsigned u = __builtin_bit_cast(unsigned, f);
  u += 0x7fffu + ((u >> 16) & 1u);   // round-to-nearest-even
  return (unsigned short)(u >> 16);
}

static __device__ __forceinline__ v8f wmma_bf16(v16bf a, v16bf b, v8f c) {
  return __builtin_amdgcn_wmma_f32_16x16x32_bf16(false, a, false, b, (short)0, c, false, false);
}

// ---- A-tile (16x32 bf16) loaders, ISA layout:
// lane l: row m=l&15, half hh=l>>4; elem j<8 -> k=8*hh+j ; j>=8 -> k=16+8*hh+(j-8)
static __device__ __forceinline__ v16bf load_a_bf16(const unsigned short* row, int kt, int lane) {
  int hh = lane >> 4;
  const unsigned short* p = row + kt * 32 + 8 * hh;
  v16u t;
  *(v4u*)&t       = *(const v4u*)p;
  *((v4u*)&t + 1) = *(const v4u*)(p + 16);
  return __builtin_bit_cast(v16bf, t);
}
static __device__ __forceinline__ v16bf load_a_f32(const float* row, int kt, int lane) {
  int hh = lane >> 4;
  const float* p = row + kt * 32 + 8 * hh;
  v16u t;
#pragma unroll
  for (int j = 0; j < 8; ++j) {
    t[j]     = f2bf_bits(p[j]);
    t[j + 8] = f2bf_bits(p[16 + j]);
  }
  return __builtin_bit_cast(v16bf, t);
}
// B-tiles pre-packed lane-major: tile*512 + lane*16 ushorts
static __device__ __forceinline__ v16bf load_b(const unsigned short* w, int tile, int lane) {
  return ((const v16bf*)w)[tile * 32 + lane];
}

// ---------------------------------------------------------------- prep kernels
__global__ void convert_x_kernel(const float* __restrict__ x, unsigned short* __restrict__ xb, long total) {
  for (long i = blockIdx.x * (long)blockDim.x + threadIdx.x; i < total; i += (long)gridDim.x * blockDim.x)
    xb[i] = f2bf_bits(x[i]);
}

// pack n_mats matrices of (K_total x N_total, row-major) into WMMA B-tile layout (bf16):
// B 32x16 tile, lane l: col n=l&15; elem j<8 -> k = 8*(l>>4) + j ; j>=8 -> k = 16 + 8*(l>>4) + (j-8)  (within 32-wide k-window)
__global__ void pack_weights_kernel(const float* __restrict__ src, unsigned short* __restrict__ dst,
                                    int K_total, int N_total, int n_mats) {
  int tiles_n = N_total >> 4;
  int per_mat = (K_total >> 5) * tiles_n * 512;
  long total  = (long)n_mats * per_mat;
  for (long t = blockIdx.x * (long)blockDim.x + threadIdx.x; t < total; t += (long)gridDim.x * blockDim.x) {
    int mat  = (int)(t / per_mat);
    int rem  = (int)(t - (long)mat * per_mat);
    int tile = rem >> 9;
    int e    = rem & 511;
    int lane = e >> 4, j = e & 15;
    int kt = tile / tiles_n, nt = tile - kt * tiles_n;
    int n  = nt * 16 + (lane & 15);
    int k  = kt * 32 + ((j >> 3) << 4) + ((lane >> 4) << 3) + (j & 7);
    dst[t] = f2bf_bits(src[((long)mat * K_total + k) * N_total + n]);
  }
}

// ---------------------------------------------------------------- QK logits
__global__ __launch_bounds__(256) void qk_logits_kernel(
    const unsigned short* __restrict__ xb, const int* __restrict__ nidx,
    const unsigned short* __restrict__ bq, const unsigned short* __restrict__ bk,
    float* __restrict__ logits) {
  const int lane = threadIdx.x & 31, wave = threadIdx.x >> 5;
  const int base = (blockIdx.x * 8 + wave) << 4;
  if (base >= NN) return;
  const int m = lane & 15;

  const unsigned short* selfrow = xb + (long)(base + m) * DD;
  v16bf ax[4];
#pragma unroll
  for (int kt = 0; kt < 4; ++kt) ax[kt] = load_a_bf16(selfrow, kt, lane);

  v8f q[8];  // [h*2 + t]
#pragma unroll
  for (int h = 0; h < 4; ++h)
#pragma unroll
    for (int t = 0; t < 2; ++t) {
      v8f c = {0.f,0.f,0.f,0.f,0.f,0.f,0.f,0.f};
#pragma unroll
      for (int kt = 0; kt < 4; ++kt) c = wmma_bf16(ax[kt], load_b(bq, (h * 4 + kt) * 2 + t, lane), c);
      q[h * 2 + t] = c;
    }

  for (int r = 0; r < 5; ++r) {
    v16bf an[4];
    if (r == 0) {
#pragma unroll
      for (int kt = 0; kt < 4; ++kt) an[kt] = ax[kt];
    } else {
      int row = nidx[(base + m) * 4 + (r - 1)];
      const unsigned short* nrow = xb + (long)row * DD;
#pragma unroll
      for (int kt = 0; kt < 4; ++kt) an[kt] = load_a_bf16(nrow, kt, lane);
    }
#pragma unroll
    for (int h = 0; h < 4; ++h) {
      v8f k0 = {0.f,0.f,0.f,0.f,0.f,0.f,0.f,0.f};
      v8f k1 = {0.f,0.f,0.f,0.f,0.f,0.f,0.f,0.f};
#pragma unroll
      for (int kt = 0; kt < 4; ++kt) {
        k0 = wmma_bf16(an[kt], load_b(bk, ((h * 5 + r) * 4 + kt) * 2 + 0, lane), k0);
        k1 = wmma_bf16(an[kt], load_b(bk, ((h * 5 + r) * 4 + kt) * 2 + 1, lane), k1);
      }
      // per-node dot over dk=32: elementwise product, reduce across the 16-lane group
      float p[8];
#pragma unroll
      for (int v = 0; v < 8; ++v) {
        float pv = q[h * 2][v] * k0[v] + q[h * 2 + 1][v] * k1[v];
#pragma unroll
        for (int off = 1; off < 16; off <<= 1) pv += __shfl_xor(pv, off, 32);
        p[v] = pv;
      }
      int vv = lane & 7;
      float a0 = (vv & 1) ? p[1] : p[0];
      float a1 = (vv & 1) ? p[3] : p[2];
      float a2 = (vv & 1) ? p[5] : p[4];
      float a3 = (vv & 1) ? p[7] : p[6];
      float b0 = (vv & 2) ? a1 : a0;
      float b1 = (vv & 2) ? a3 : a2;
      float lg = ((vv & 4) ? b1 : b0) * QK_SCALE;
      if ((lane & 15) < 8) {
        int node = base + vv + ((lane >> 4) << 3);
        logits[(long)(h * 5 + r) * NN + node] = lg;
      }
    }
  }
}

// ---------------------------------------------------------------- softmax reductions (over node axis)
__global__ __launch_bounds__(256) void col_max_kernel(const float* __restrict__ logits, float* __restrict__ smax) {
  __shared__ float red[256];
  const float* col = logits + (long)blockIdx.x * NN;
  float mx = -3.4e38f;
  for (int i = threadIdx.x; i < NN; i += 256) mx = fmaxf(mx, col[i]);
  red[threadIdx.x] = mx;
  __syncthreads();
  for (int s = 128; s > 0; s >>= 1) {
    if (threadIdx.x < s) red[threadIdx.x] = fmaxf(red[threadIdx.x], red[threadIdx.x + s]);
    __syncthreads();
  }
  if (threadIdx.x == 0) smax[blockIdx.x] = red[0];
}

__global__ __launch_bounds__(256) void col_sum_kernel(const float* __restrict__ logits,
                                                      const float* __restrict__ smax,
                                                      float* __restrict__ ssum) {
  __shared__ float red[256];
  const float* col = logits + (long)blockIdx.x * NN;
  float mx = smax[blockIdx.x];
  float s = 0.f;
  for (int i = threadIdx.x; i < NN; i += 256) s += __expf(col[i] - mx);
  red[threadIdx.x] = s;
  __syncthreads();
  for (int t = 128; t > 0; t >>= 1) {
    if (threadIdx.x < t) red[threadIdx.x] += red[threadIdx.x + t];
    __syncthreads();
  }
  if (threadIdx.x == 0) ssum[blockIdx.x] = red[0];
}

// ---------------------------------------------------------------- attention + LN1 (4 waves / block)
__global__ __launch_bounds__(128) void attn_ln1_kernel(
    const unsigned short* __restrict__ xb, const float* __restrict__ x, const int* __restrict__ nidx,
    const unsigned short* __restrict__ bv, const float* __restrict__ logits,
    const float* __restrict__ smax, const float* __restrict__ ssum,
    const float* __restrict__ g1, const float* __restrict__ be1, float* __restrict__ h1) {
  __shared__ float wlds[4][320];       // softmax weights per wave: [hr][m]
  __shared__ float stage[4][16 * DD];  // attn tile per wave
  const int lane = threadIdx.x & 31, wave = threadIdx.x >> 5;
  const int base = (blockIdx.x * 4 + wave) << 4;
  if (base >= NN) return;
  const int m = lane & 15, hh = lane >> 4;

  for (int i = lane; i < 320; i += 32) {
    int hr = i >> 4, mm = i & 15;
    float lg = logits[(long)hr * NN + base + mm];
    wlds[wave][i] = __expf(lg - smax[hr]) / ssum[hr];
  }
  asm volatile("s_wait_dscnt 0" ::: "memory");

  v8f acc[8];  // [h*2 + t]
#pragma unroll
  for (int i = 0; i < 8; ++i) acc[i] = (v8f){0.f,0.f,0.f,0.f,0.f,0.f,0.f,0.f};

  for (int r = 0; r < 5; ++r) {
    const unsigned short* nrow;
    if (r == 0) nrow = xb + (long)(base + m) * DD;
    else        nrow = xb + (long)nidx[(base + m) * 4 + (r - 1)] * DD;
    v16bf an[4];
#pragma unroll
    for (int kt = 0; kt < 4; ++kt) an[kt] = load_a_bf16(nrow, kt, lane);
#pragma unroll
    for (int h = 0; h < 4; ++h) {
      v8f v0 = {0.f,0.f,0.f,0.f,0.f,0.f,0.f,0.f};
      v8f v1 = {0.f,0.f,0.f,0.f,0.f,0.f,0.f,0.f};
#pragma unroll
      for (int kt = 0; kt < 4; ++kt) {
        v0 = wmma_bf16(an[kt], load_b(bv, ((h * 5 + r) * 4 + kt) * 2 + 0, lane), v0);
        v1 = wmma_bf16(an[kt], load_b(bv, ((h * 5 + r) * 4 + kt) * 2 + 1, lane), v1);
      }
#pragma unroll
      for (int v = 0; v < 8; ++v) {
        float w = wlds[wave][(h * 5 + r) * 16 + v + (hh << 3)];
        acc[h * 2][v]     += w * v0[v];
        acc[h * 2 + 1][v] += w * v1[v];
      }
    }
  }
  // stage attn: element (row = v+8*hh, col = h*32 + t*16 + m)
#pragma unroll
  for (int h = 0; h < 4; ++h)
#pragma unroll
    for (int t = 0; t < 2; ++t)
#pragma unroll
      for (int v = 0; v < 8; ++v)
        stage[wave][(v + (hh << 3)) * DD + h * 32 + t * 16 + m] = acc[h * 2 + t][v];
  asm volatile("s_wait_dscnt 0" ::: "memory");

  // residual + LayerNorm, one node row per iteration, fully coalesced
  float4 g4  = *(const float4*)(g1 + lane * 4);
  float4 bt4 = *(const float4*)(be1 + lane * 4);
  for (int mm = 0; mm < 16; ++mm) {
    long node = base + mm;
    float4 xv = *(const float4*)(x + node * DD + lane * 4);
    const float* sp = &stage[wave][mm * DD + lane * 4];
    float y0 = xv.x + sp[0], y1 = xv.y + sp[1], y2 = xv.z + sp[2], y3 = xv.w + sp[3];
    float s  = y0 + y1 + y2 + y3;
    float ss = y0 * y0 + y1 * y1 + y2 * y2 + y3 * y3;
#pragma unroll
    for (int off = 1; off < 32; off <<= 1) { s += __shfl_xor(s, off, 32); ss += __shfl_xor(ss, off, 32); }
    float mu  = s * (1.0f / 128.0f);
    float var = ss * (1.0f / 128.0f) - mu * mu;
    float inv = rsqrtf(var + EPSL);
    float4 o;
    o.x = (y0 - mu) * inv * g4.x + bt4.x;
    o.y = (y1 - mu) * inv * g4.y + bt4.y;
    o.z = (y2 - mu) * inv * g4.z + bt4.z;
    o.w = (y3 - mu) * inv * g4.w + bt4.w;
    *(float4*)(h1 + node * DD + lane * 4) = o;
  }
}

// ---------------------------------------------------------------- FFN + LN2 (4 waves / block, 64KB LDS)
__global__ __launch_bounds__(128) void ffn_ln2_kernel(
    const float* __restrict__ h1, const unsigned short* __restrict__ bW1, const unsigned short* __restrict__ bW2,
    const float* __restrict__ b1, const float* __restrict__ b2,
    const float* __restrict__ g2, const float* __restrict__ be2, float* __restrict__ out) {
  __shared__ unsigned short ffs[4][16 * DFFN];  // 16KB per wave
  const int lane = threadIdx.x & 31, wave = threadIdx.x >> 5;
  const int base = (blockIdx.x * 4 + wave) << 4;
  if (base >= NN) return;
  const int m = lane & 15, hh = lane >> 4;

  const float* hrow = h1 + (long)(base + m) * DD;
  v16bf ah[4];
#pragma unroll
  for (int kt = 0; kt < 4; ++kt) ah[kt] = load_a_f32(hrow, kt, lane);

  // GEMM1: (16x128) @ (128x512) + b1, relu -> LDS (bf16, row-major per wave)
  for (int nt = 0; nt < 32; ++nt) {
    float bias = b1[nt * 16 + m];
    v8f c = {bias, bias, bias, bias, bias, bias, bias, bias};
#pragma unroll
    for (int kt = 0; kt < 4; ++kt) c = wmma_bf16(ah[kt], load_b(bW1, kt * 32 + nt, lane), c);
#pragma unroll
    for (int v = 0; v < 8; ++v)
      ffs[wave][(v + (hh << 3)) * DFFN + nt * 16 + m] = f2bf_bits(fmaxf(c[v], 0.f));
  }
  asm volatile("s_wait_dscnt 0" ::: "memory");

  // GEMM2: (16x512) @ (512x128) + b2
  v8f acc2[8];
#pragma unroll
  for (int nt2 = 0; nt2 < 8; ++nt2) {
    float bias = b2[nt2 * 16 + m];
    acc2[nt2] = (v8f){bias, bias, bias, bias, bias, bias, bias, bias};
  }
  for (int kt2 = 0; kt2 < 16; ++kt2) {
    const unsigned short* sp = &ffs[wave][m * DFFN + kt2 * 32 + 8 * hh];
    v16u t;
    *(v4u*)&t       = *(const v4u*)sp;
    *((v4u*)&t + 1) = *(const v4u*)(sp + 16);
    v16bf a2 = __builtin_bit_cast(v16bf, t);
#pragma unroll
    for (int nt2 = 0; nt2 < 8; ++nt2)
      acc2[nt2] = wmma_bf16(a2, load_b(bW2, kt2 * 8 + nt2, lane), acc2[nt2]);
  }

  // stage ffn output (f32) into the (now dead) front of this wave's LDS region
  float* lns = (float*)&ffs[wave][0];  // 16*128 f32 = 8KB <= 16KB
#pragma unroll
  for (int nt2 = 0; nt2 < 8; ++nt2)
#pragma unroll
    for (int v = 0; v < 8; ++v)
      lns[(v + (hh << 3)) * DD + nt2 * 16 + m] = acc2[nt2][v];
  asm volatile("s_wait_dscnt 0" ::: "memory");

  // residual + LayerNorm2 -> out
  float4 g4  = *(const float4*)(g2 + lane * 4);
  float4 bt4 = *(const float4*)(be2 + lane * 4);
  for (int mm = 0; mm < 16; ++mm) {
    long node = base + mm;
    float4 hv = *(const float4*)(h1 + node * DD + lane * 4);
    const float* sp = &lns[mm * DD + lane * 4];
    float y0 = hv.x + sp[0], y1 = hv.y + sp[1], y2 = hv.z + sp[2], y3 = hv.w + sp[3];
    float s  = y0 + y1 + y2 + y3;
    float ss = y0 * y0 + y1 * y1 + y2 * y2 + y3 * y3;
#pragma unroll
    for (int off = 1; off < 32; off <<= 1) { s += __shfl_xor(s, off, 32); ss += __shfl_xor(ss, off, 32); }
    float mu  = s * (1.0f / 128.0f);
    float var = ss * (1.0f / 128.0f) - mu * mu;
    float inv = rsqrtf(var + EPSL);
    float4 o;
    o.x = (y0 - mu) * inv * g4.x + bt4.x;
    o.y = (y1 - mu) * inv * g4.y + bt4.y;
    o.z = (y2 - mu) * inv * g4.z + bt4.z;
    o.w = (y3 - mu) * inv * g4.w + bt4.w;
    *(float4*)(out + node * DD + lane * 4) = o;
  }
}

// ---------------------------------------------------------------- host
extern "C" void kernel_launch(void* const* d_in, const int* in_sizes, int n_in,
                              void* d_out, int out_size, void* d_ws, size_t ws_size,
                              hipStream_t stream) {
  (void)in_sizes; (void)n_in; (void)out_size; (void)ws_size;
  const float* x   = (const float*)d_in[0];
  const int*   nid = (const int*)d_in[1];
  const float* w_q = (const float*)d_in[2];
  const float* w_k = (const float*)d_in[3];
  const float* w_v = (const float*)d_in[4];
  const float* W1  = (const float*)d_in[5];
  const float* b1  = (const float*)d_in[6];
  const float* W2  = (const float*)d_in[7];
  const float* b2  = (const float*)d_in[8];
  const float* g1  = (const float*)d_in[9];
  const float* be1 = (const float*)d_in[10];
  const float* g2  = (const float*)d_in[11];
  const float* be2 = (const float*)d_in[12];
  float* out = (float*)d_out;

  char* ws = (char*)d_ws;
  size_t off = 0;
  auto carve = [&](size_t bytes) -> void* {
    off = (off + 255) & ~(size_t)255;
    void* p = ws + off;
    off += bytes;
    return p;
  };
  unsigned short* xb  = (unsigned short*)carve((size_t)NN * DD * 2);
  unsigned short* bq  = (unsigned short*)carve((size_t)4 * 128 * 32 * 2);
  unsigned short* bk  = (unsigned short*)carve((size_t)20 * 128 * 32 * 2);
  unsigned short* bv  = (unsigned short*)carve((size_t)20 * 128 * 32 * 2);
  unsigned short* bW1 = (unsigned short*)carve((size_t)128 * 512 * 2);
  unsigned short* bW2 = (unsigned short*)carve((size_t)512 * 128 * 2);
  float* logits = (float*)carve((size_t)20 * NN * 4);
  float* smax   = (float*)carve(20 * 4);
  float* ssum   = (float*)carve(20 * 4);
  float* h1     = (float*)carve((size_t)NN * DD * 4);

  convert_x_kernel<<<2048, 256, 0, stream>>>(x, xb, (long)NN * DD);
  pack_weights_kernel<<<64, 256, 0, stream>>>(w_q, bq, 128, 32, 4);
  pack_weights_kernel<<<320, 256, 0, stream>>>(w_k, bk, 128, 32, 20);
  pack_weights_kernel<<<320, 256, 0, stream>>>(w_v, bv, 128, 32, 20);
  pack_weights_kernel<<<256, 256, 0, stream>>>(W1, bW1, 128, 512, 1);
  pack_weights_kernel<<<256, 256, 0, stream>>>(W2, bW2, 512, 128, 1);

  const int waves = NN / 16;                 // 6250
  qk_logits_kernel<<<(waves + 7) / 8, 256, 0, stream>>>(xb, nid, bq, bk, logits);
  col_max_kernel<<<20, 256, 0, stream>>>(logits, smax);
  col_sum_kernel<<<20, 256, 0, stream>>>(logits, smax, ssum);
  attn_ln1_kernel<<<(waves + 3) / 4, 128, 0, stream>>>(xb, x, nid, bv, logits, smax, ssum, g1, be1, h1);
  ffn_ln2_kernel<<<(waves + 3) / 4, 128, 0, stream>>>(h1, bW1, bW2, b1, b2, g2, be2, out);
}
